// DiSA_39831526703536
// MI455X (gfx1250) — matrix operand order repaired and checked by reading, not compiled, not run on hardware
//
#include <hip/hip_runtime.h>
#include <hip/hip_bf16.h>

typedef __attribute__((ext_vector_type(16))) _Float16 v16h;
typedef __attribute__((ext_vector_type(8)))  _Float16 v8h;
typedef __attribute__((ext_vector_type(8)))  float    v8f;

#define D_E   300
#define D_EP  320                 /* padded K for fc gemm */
#define D_H   128
#define SEQ   512
#define BATCH 4
#define M_TOT (BATCH * SEQ)       /* 2048 rows */
#define CCLAMP 5.0f

/* ---- workspace layout (bytes), all 16B aligned ---- */
#define OFF_REP_F32   0
#define OFF_PK        (OFF_REP_F32  + M_TOT * D_H * 4)      /* float2 {dep, rep} */
#define OFF_HEAD      (OFF_PK       + M_TOT * D_H * 8)
#define OFF_G1        (OFF_HEAD     + M_TOT * D_H * 4)
#define OFF_ATTN_F32  (OFF_G1       + M_TOT * D_H * 4)
#define OFF_REP_F16   (OFF_ATTN_F32 + M_TOT * D_H * 4)
#define OFF_ATTN_F16  (OFF_REP_F16  + M_TOT * D_H * 2)
#define OFF_XH        (OFF_ATTN_F16 + M_TOT * D_H * 2)
#define OFF_FWH       (OFF_XH       + M_TOT * D_EP * 2)
#define OFF_W1H       (OFF_FWH      + D_H * D_EP * 2)
#define OFF_W2H       (OFF_W1H      + D_H * D_H * 2)
#define OFF_WF1H      (OFF_W2H      + D_H * D_H * 2)
#define OFF_WF2H      (OFF_WF1H     + D_H * D_H * 2)

static __device__ __forceinline__ v16h cat16(v8h lo, v8h hi) {
    return __builtin_shufflevector(lo, hi, 0,1,2,3,4,5,6,7,8,9,10,11,12,13,14,15);
}

static __device__ __forceinline__ float fast_tanh(float x) {
#if __has_builtin(__builtin_amdgcn_tanhf)
    return __builtin_amdgcn_tanhf(x);
#else
    float r;
    asm volatile("v_tanh_f32 %0, %1\n\tv_nop" : "=v"(r) : "v"(x));
    return r;
#endif
}

/* =====================================================================
 * Prep: f32 -> f16 convert with zero K-padding (coalesced).
 * =====================================================================*/
__global__ __launch_bounds__(256)
void k_cvt_pad(const float* __restrict__ src, _Float16* __restrict__ dst,
               int rows, int K, int Kp)
{
    int idx = blockIdx.x * 256 + threadIdx.x;
    if (idx >= rows * Kp) return;
    int r = idx / Kp, k = idx - r * Kp;
    dst[idx] = (k < K) ? (_Float16)src[r * K + k] : (_Float16)0.f;
}

/* =====================================================================
 * Kernel 1: rep_map = elu(xh @ fwh^T + fc_b). M=2048,N=128,K=320.
 * One wave per 16x16 tile; unconditional b128 fragment loads.
 * =====================================================================*/
__global__ __launch_bounds__(128)
void k_fc_elu(const _Float16* __restrict__ xh, const _Float16* __restrict__ fwh,
              const float* __restrict__ fc_b,
              float* __restrict__ rep_f32, _Float16* __restrict__ rep_f16)
{
    const int lane = threadIdx.x & 31;
    const int wid  = threadIdx.x >> 5;
    const int tile = blockIdx.x * 4 + wid;     // 0..1023
    const int tm = tile >> 3, tn = tile & 7;
    const int hi = lane >> 4, ln = lane & 15;

    const _Float16* ap = xh  + (tm * 16 + ln) * D_EP;
    const _Float16* bp = fwh + (tn * 16 + ln) * D_EP;

    v8f c = {};
#pragma unroll
    for (int k0 = 0; k0 < D_EP; k0 += 32) {
        v16h a  = cat16(*(const v8h*)(ap + k0 + hi * 8),
                        *(const v8h*)(ap + k0 + 16 + hi * 8));
        v16h bf = cat16(*(const v8h*)(bp + k0 + hi * 8),
                        *(const v8h*)(bp + k0 + 16 + hi * 8));
        c = __builtin_amdgcn_wmma_f32_16x16x32_f16(false, a, false, bf,
                                                   (short)0, c, false, false);
    }
#pragma unroll
    for (int r = 0; r < 8; ++r) {
        int gm = tm * 16 + hi * 8 + r;
        int gn = tn * 16 + ln;
        float v = c[r] + fc_b[gn];
        v = (v > 0.f) ? v : (expf(v) - 1.f);   // elu, alpha=1
        rep_f32[gm * D_H + gn] = v;
        rep_f16[gm * D_H + gn] = (_Float16)v;
    }
}

/* K=128 f16 WMMA tile: A row pointer / B col pointer pre-offset. */
static __device__ __forceinline__ v8f gemm_k128(const _Float16* __restrict__ ap,
                                                const _Float16* __restrict__ bp,
                                                int hi)
{
    v8f c = {};
#pragma unroll
    for (int k0 = 0; k0 < D_H; k0 += 32) {
        v16h a  = cat16(*(const v8h*)(ap + k0 + hi * 8),
                        *(const v8h*)(ap + k0 + 16 + hi * 8));
        v16h bf = cat16(*(const v8h*)(bp + k0 + hi * 8),
                        *(const v8h*)(bp + k0 + 16 + hi * 8));
        c = __builtin_amdgcn_wmma_f32_16x16x32_f16(false, a, false, bf,
                                                   (short)0, c, false, false);
    }
    return c;
}

/* =====================================================================
 * Kernel 2: y=0: dep (packed with rep as float2), y=1: head, y=2: g1.
 * =====================================================================*/
__global__ __launch_bounds__(128)
void k_gemm3(const _Float16* __restrict__ reph,
             const _Float16* __restrict__ w1h, const _Float16* __restrict__ w2h,
             const _Float16* __restrict__ wf1h,
             const float* __restrict__ rep_f32,
             float2* __restrict__ pk, float* __restrict__ head,
             float* __restrict__ g1)
{
    const _Float16* W = (blockIdx.y == 0) ? w1h : (blockIdx.y == 1) ? w2h : wf1h;

    const int lane = threadIdx.x & 31;
    const int wid  = threadIdx.x >> 5;
    const int tile = blockIdx.x * 4 + wid;
    const int tm = tile >> 3, tn = tile & 7;
    const int hi = lane >> 4, ln = lane & 15;

    v8f c = gemm_k128(reph + (tm * 16 + ln) * D_H, W + (tn * 16 + ln) * D_H, hi);

#pragma unroll
    for (int r = 0; r < 8; ++r) {
        int gm  = tm * 16 + hi * 8 + r;
        int gn  = tn * 16 + ln;
        int idx = gm * D_H + gn;
        if (blockIdx.y == 0)      pk[idx]  = make_float2(c[r], rep_f32[idx]);
        else if (blockIdx.y == 1) head[idx] = c[r];
        else                      g1[idx]   = c[r];
    }
}

/* =====================================================================
 * Kernel 3: fused masked softmax + weighted sum over dependent axis j.
 * Clamp C bounds logits <= C, and softmax is shift-invariant, so use the
 * constant shift exp(l - C): one transcendental pair (tanh+exp) per
 * element, branchless 0/1 mask multiply. Matches reference semantics
 * (including the all-masked row -> 0 guard).
 * =====================================================================*/
__global__ __launch_bounds__(128)
void k_attn(const float2* __restrict__ pk, const float* __restrict__ head,
            const float* __restrict__ b_1, const int* __restrict__ rep_mask,
            float* __restrict__ attn_f32, _Float16* __restrict__ attn_f16)
{
    const int i = blockIdx.x;
    const int b = blockIdx.y;
    const int h = threadIdx.x;
    const int base = b * SEQ;

    const float hv   = head[(base + i) * D_H + h] + b_1[h];
    const float invc = 1.0f / CCLAMP;

    float s = 0.f, acc = 0.f;
    for (int j = i + 1; j < SEQ; ++j) {
        float2 dr  = pk[(base + j) * D_H + h];           // {dep, rep}
        float  msk = (float)rep_mask[base + j];          // 0/1
        float  l   = CCLAMP * fast_tanh((dr.x + hv) * invc);
        float  e   = __expf(l - CCLAMP) * msk;           // shift by clamp bound
        s   += e;
        acc += e * dr.y;
    }
    float r = (s > 0.f) ? (acc / s) : 0.f;               // zero-sum guard
    attn_f32[(base + i) * D_H + h] = r;
    attn_f16[(base + i) * D_H + h] = (_Float16)r;
}

/* =====================================================================
 * Kernel 4: g2 = attn @ wf2^T, fused sigmoid gate + blend + row mask.
 * =====================================================================*/
__global__ __launch_bounds__(128)
void k_gate_out(const _Float16* __restrict__ attnh, const _Float16* __restrict__ wf2h,
                const float* __restrict__ b_f, const float* __restrict__ g1,
                const float* __restrict__ rep_f32, const float* __restrict__ attn_f32,
                const int* __restrict__ rep_mask, float* __restrict__ out)
{
    const int lane = threadIdx.x & 31;
    const int wid  = threadIdx.x >> 5;
    const int tile = blockIdx.x * 4 + wid;
    const int tm = tile >> 3, tn = tile & 7;
    const int hi = lane >> 4, ln = lane & 15;

    v8f c = gemm_k128(attnh + (tm * 16 + ln) * D_H, wf2h + (tn * 16 + ln) * D_H, hi);

#pragma unroll
    for (int r = 0; r < 8; ++r) {
        int gm  = tm * 16 + hi * 8 + r;
        int gn  = tn * 16 + ln;
        int idx = gm * D_H + gn;
        float g    = c[r] + g1[idx] + b_f[gn];
        float gate = 1.f / (1.f + __expf(-g));
        float o    = gate * rep_f32[idx] + (1.f - gate) * attn_f32[idx];
        out[idx]   = o * (float)rep_mask[gm];
    }
}

extern "C" void kernel_launch(void* const* d_in, const int* in_sizes, int n_in,
                              void* d_out, int out_size, void* d_ws, size_t ws_size,
                              hipStream_t stream)
{
    (void)in_sizes; (void)n_in; (void)out_size; (void)ws_size;

    const float* x     = (const float*)d_in[0];
    const int*   rmask = (const int*)  d_in[1];
    const float* fc_w  = (const float*)d_in[2];
    const float* fc_b  = (const float*)d_in[3];
    const float* w1_w  = (const float*)d_in[4];
    const float* w2_w  = (const float*)d_in[5];
    const float* b_1   = (const float*)d_in[6];
    const float* wf1_w = (const float*)d_in[7];
    const float* wf2_w = (const float*)d_in[8];
    const float* b_f   = (const float*)d_in[9];

    char* ws = (char*)d_ws;
    float*    rep_f32  = (float*)   (ws + OFF_REP_F32);
    float2*   pk       = (float2*)  (ws + OFF_PK);
    float*    head     = (float*)   (ws + OFF_HEAD);
    float*    g1       = (float*)   (ws + OFF_G1);
    float*    attn_f32 = (float*)   (ws + OFF_ATTN_F32);
    _Float16* rep_f16  = (_Float16*)(ws + OFF_REP_F16);
    _Float16* attn_f16 = (_Float16*)(ws + OFF_ATTN_F16);
    _Float16* xh       = (_Float16*)(ws + OFF_XH);
    _Float16* fwh      = (_Float16*)(ws + OFF_FWH);
    _Float16* w1h      = (_Float16*)(ws + OFF_W1H);
    _Float16* w2h      = (_Float16*)(ws + OFF_W2H);
    _Float16* wf1h     = (_Float16*)(ws + OFF_WF1H);
    _Float16* wf2h     = (_Float16*)(ws + OFF_WF2H);

    /* ---- f16 conversion / padding ---- */
    k_cvt_pad<<<(M_TOT * D_EP + 255) / 256, 256, 0, stream>>>(x,     xh,  M_TOT, D_E, D_EP);
    k_cvt_pad<<<(D_H   * D_EP + 255) / 256, 256, 0, stream>>>(fc_w,  fwh, D_H,   D_E, D_EP);
    k_cvt_pad<<<(D_H   * D_H  + 255) / 256, 256, 0, stream>>>(w1_w,  w1h,  D_H, D_H, D_H);
    k_cvt_pad<<<(D_H   * D_H  + 255) / 256, 256, 0, stream>>>(w2_w,  w2h,  D_H, D_H, D_H);
    k_cvt_pad<<<(D_H   * D_H  + 255) / 256, 256, 0, stream>>>(wf1_w, wf1h, D_H, D_H, D_H);
    k_cvt_pad<<<(D_H   * D_H  + 255) / 256, 256, 0, stream>>>(wf2_w, wf2h, D_H, D_H, D_H);

    /* ---- compute pipeline ---- */
    k_fc_elu  <<<256,              128, 0, stream>>>(xh, fwh, fc_b, rep_f32, rep_f16);
    k_gemm3   <<<dim3(256, 3),     128, 0, stream>>>(rep_f16, w1h, w2h, wf1h,
                                                     rep_f32, pk, head, g1);
    k_attn    <<<dim3(SEQ, BATCH), 128, 0, stream>>>(pk, head, b_1, rmask,
                                                     attn_f32, attn_f16);
    k_gate_out<<<256,              128, 0, stream>>>(attn_f16, wf2h, b_f, g1,
                                                     rep_f32, attn_f32, rmask,
                                                     (float*)d_out);
}
